// CoLESEncoder_78451872628885
// MI455X (gfx1250) — matrix operand order repaired and checked by reading, not compiled, not run on hardware
//
#include <hip/hip_runtime.h>
#include <math.h>

// Problem constants (from reference)
constexpr int BB = 64;     // batch
constexpr int TT = 2048;   // time
constexpr int II = 128;    // input dim
constexpr int HH = 128;    // hidden
constexpr int GG = 384;    // 3H
constexpr int EE = 64;     // proj out

typedef __attribute__((ext_vector_type(16))) __bf16 v16bf;
typedef __attribute__((ext_vector_type(8)))  __bf16 v8bf;
typedef __attribute__((ext_vector_type(8)))  float  v8f;
typedef __attribute__((ext_vector_type(4)))  float  v4f;

union Frag16 { v16bf v; __bf16 e[16]; };

// Flat pointers into __shared__ are {SHARED_BASE_hi32, lds_offset32}; the HW
// truncates to addr[31:0] for the LDS address, so the low 32 bits are the
// DS offset usable by async-load-to-LDS.
__device__ __forceinline__ unsigned ldsOff(const void* p) {
  return (unsigned)(uintptr_t)p;
}

// CDNA5 async global->LDS copy (tracked with ASYNCcnt). GV mode: 64-bit vaddr.
__device__ __forceinline__ void asyncLdB128(unsigned lds, const void* gptr) {
  asm volatile("global_load_async_to_lds_b128 %0, %1, off"
               :: "v"(lds), "v"((unsigned long long)(uintptr_t)gptr)
               : "memory");
}
#define WAIT_ASYNC(n) asm volatile("s_wait_asynccnt " #n ::: "memory")

// ---------------------------------------------------------------------------
// WMMA fragment layout (CDNA5 ISA 7.12.2, wave32):
//   A (16x32 bf16): lane m = lane&15; kb = (lane>>4)*8;
//       elems 0..7 -> K=kb+0..7 ; elems 8..15 -> K=kb+16..23
//   B (32x16 bf16): lane n = lane&15; kb = (lane>>4)*8; same elem->K map as A
//   C/D (16x16 f32): elem r -> row = r + (lane>>4)*8, col = lane&15
// ---------------------------------------------------------------------------

// Phase 1: gi[t][b][g] = x[b][t][:] . w_ih[g][:] + b_ih[g]
// Async double-buffered x tiles; w_ih fragments resident in VGPRs.
__global__ __launch_bounds__(256) void gi_gemm_kernel(
    const float* __restrict__ x, const float* __restrict__ w_ih,
    const float* __restrict__ b_ih, float* __restrict__ gi, int numWG) {
  __shared__ __align__(16) float xt[2][16 * II];  // 2 x 8 KB double buffer

  const int tid  = threadIdx.x;
  const int lane = tid & 31;
  const int wave = tid >> 5;          // 0..7 -> g-tiles wave*3 .. wave*3+2

  const int n   = lane & 15;
  const int kbB = (lane >> 4) * 8;
  const int m   = lane & 15;
  const int kbA = (lane >> 4) * 8;
  const int mhi = (lane >> 4) * 8;

  // Resident w_ih fragments (bf16) + bias folded into accumulator init
  Frag16 bfrag[3][4];
  float  bias[3];
  for (int gt = 0; gt < 3; ++gt) {
    const int g = (wave * 3 + gt) * 16 + n;
    const float* wr = w_ih + (size_t)g * II;
    for (int kc = 0; kc < 4; ++kc)
      for (int j = 0; j < 16; ++j)
        bfrag[gt][kc].e[j] = (__bf16)wr[kc * 32 + kbB + (j & 7) + (j >> 3) * 16];
    bias[gt] = b_ih[g];
  }

  auto issueTile = [&](int rt, int buf) {
    const char* src = (const char*)(x + (size_t)rt * 16 * II);  // 8 KB contig
    const unsigned dst = ldsOff(&xt[buf][0]);
    for (int r = 0; r < 2; ++r) {                 // 256 thr x 2 x 16B = 8 KB
      const int off = tid * 16 + r * 4096;
      asyncLdB128(dst + off, src + off);
    }
  };

  const int nRowTiles = (BB * TT) / 16;  // 8192
  issueTile(blockIdx.x, 0);              // every block has >=1 tile (512<<8192)

  int buf = 0;
  for (int rt = blockIdx.x; rt < nRowTiles; rt += numWG, buf ^= 1) {
    const bool hasNext = (rt + numWG) < nRowTiles;   // uniform per block
    if (hasNext) {
      issueTile(rt + numWG, buf ^ 1);   // overlap next tile with this compute
      WAIT_ASYNC(0x2);                  // 2 newer outstanding -> current done
    } else {
      WAIT_ASYNC(0x0);
    }
    __syncthreads();

    // A fragments: 4x b128 LDS loads + cvt per k-chunk
    Frag16 afrag[4];
    for (int kc = 0; kc < 4; ++kc) {
      const float* p = &xt[buf][m * II + kc * 32 + kbA];
      v4f a0 = *(const v4f*)p;
      v4f a1 = *(const v4f*)(p + 4);
      v4f a2 = *(const v4f*)(p + 16);
      v4f a3 = *(const v4f*)(p + 20);
      for (int j = 0; j < 4; ++j) {
        afrag[kc].e[j]      = (__bf16)a0[j];
        afrag[kc].e[4 + j]  = (__bf16)a1[j];
        afrag[kc].e[8 + j]  = (__bf16)a2[j];
        afrag[kc].e[12 + j] = (__bf16)a3[j];
      }
    }

    const int b  = (rt * 16) / TT;   // tile never crosses batch (TT%16==0)
    const int t0 = (rt * 16) % TT;
    for (int gt = 0; gt < 3; ++gt) {
      v8f c;
      for (int r = 0; r < 8; ++r) c[r] = bias[gt];
      for (int kc = 0; kc < 4; ++kc)
        c = __builtin_amdgcn_wmma_f32_16x16x32_bf16(
            false, afrag[kc].v, false, bfrag[gt][kc].v, (short)0, c, false, false);
      const int g = (wave * 3 + gt) * 16 + n;
      for (int r = 0; r < 8; ++r)
        gi[((size_t)(t0 + r + mhi) * BB + b) * GG + g] = c[r];
    }
    __syncthreads();  // all reads of xt[buf] done before it is re-targeted
  }
}

// Phase 2: sequential GRU scan. 4 blocks x 16 batch rows. w_hh fragments in
// registers; h kept bf16 (WMMA A) + f32 (exact gates) in LDS; gi stream is
// async double-buffered into LDS so no HBM latency sits on the serial chain.
__global__ __launch_bounds__(256) void gru_scan_kernel(
    const float* __restrict__ gi, const float* __restrict__ w_hh,
    const float* __restrict__ b_hh, float* __restrict__ pooled) {
  __shared__ __align__(16) __bf16 hbf[16 * HH];      // 4 KB
  __shared__ __align__(16) float  hfs[16 * HH];      // 8 KB
  __shared__ __align__(16) float  ghs[16 * GG];      // 24 KB
  __shared__ __align__(16) float  gib[2][16 * GG];   // 2 x 24 KB gi buffers

  const int tid  = threadIdx.x;
  const int lane = tid & 31;
  const int wave = tid >> 5;
  const int b0   = blockIdx.x * 16;

  const int n   = lane & 15;
  const int kbB = (lane >> 4) * 8;
  const int m   = lane & 15;
  const int kbA = (lane >> 4) * 8;
  const int mhi = (lane >> 4) * 8;

  // Resident w_hh fragments
  Frag16 bfrag[3][4];
  float  bias[3];
  for (int gt = 0; gt < 3; ++gt) {
    const int g = (wave * 3 + gt) * 16 + n;
    const float* wr = w_hh + (size_t)g * HH;
    for (int kc = 0; kc < 4; ++kc)
      for (int j = 0; j < 16; ++j)
        bfrag[gt][kc].e[j] = (__bf16)wr[kc * 32 + kbB + (j & 7) + (j >> 3) * 16];
    bias[gt] = b_hh[g];
  }

  auto issueGi = [&](int t, int buf) {
    // gi rows (t*64+b0 .. +15) are contiguous: 16*384 f32 = 24 KB
    const char* src = (const char*)(gi + ((size_t)t * BB + b0) * GG);
    const unsigned dst = ldsOff(&gib[buf][0]);
    for (int r = 0; r < 6; ++r) {                 // 256 thr x 6 x 16B = 24 KB
      const int off = tid * 16 + r * 4096;
      asyncLdB128(dst + off, src + off);
    }
  };

  for (int idx = tid; idx < 16 * HH; idx += 256) { hbf[idx] = (__bf16)0.f; hfs[idx] = 0.f; }
  issueGi(0, 0);
  __syncthreads();

  // Elementwise ownership: row = tid>>4, 8 contiguous cols per thread
  const int erow = tid >> 4;
  const int ej0  = (tid & 15) * 8;
  float acc[8];
  for (int j = 0; j < 8; ++j) acc[j] = 0.f;

  for (int t = 0; t < TT; ++t) {
    const int cur = t & 1;
    const bool hasNext = (t + 1) < TT;            // uniform
    if (hasNext) issueGi(t + 1, cur ^ 1);         // overlap with WMMA below

    // A fragments from current h (bf16 in LDS)
    Frag16 afrag[4];
    for (int kc = 0; kc < 4; ++kc) {
      const __bf16* p = &hbf[m * HH + kc * 32 + kbA];
      v8bf lo = *(const v8bf*)p;
      v8bf hi = *(const v8bf*)(p + 16);
      for (int j = 0; j < 8; ++j) { afrag[kc].e[j] = lo[j]; afrag[kc].e[8 + j] = hi[j]; }
    }
    // gh = h @ w_hh^T + b_hh  (3 g-tiles per wave, 12 WMMA/step/wave)
    for (int gt = 0; gt < 3; ++gt) {
      v8f c;
      for (int r = 0; r < 8; ++r) c[r] = bias[gt];
      for (int kc = 0; kc < 4; ++kc)
        c = __builtin_amdgcn_wmma_f32_16x16x32_bf16(
            false, afrag[kc].v, false, bfrag[gt][kc].v, (short)0, c, false, false);
      const int g = (wave * 3 + gt) * 16 + n;
      for (int r = 0; r < 8; ++r)
        ghs[(r + mhi) * GG + g] = c[r];
    }

    // Current gi buffer is ready once the 6 newer (t+1) ops are the only
    // outstanding ones (async loads complete in order).
    if (hasNext) WAIT_ASYNC(0x6); else WAIT_ASYNC(0x0);
    __syncthreads();

    // Gates in f32, all operands now in LDS
    const float* girow = &gib[cur][erow * GG];
    for (int j = 0; j < 8; ++j) {
      const int col = ej0 + j;
      const float gr = girow[col]          + ghs[erow * GG + col];
      const float gz = girow[HH + col]     + ghs[erow * GG + HH + col];
      const float hn = ghs[erow * GG + 2 * HH + col];
      const float r  = 1.f / (1.f + __expf(-gr));
      const float z  = 1.f / (1.f + __expf(-gz));
      const float nn = tanhf(girow[2 * HH + col] + r * hn);
      const float ho = hfs[erow * HH + col];
      const float hw = (1.f - z) * nn + z * ho;
      acc[j] += hw;
      hfs[erow * HH + col] = hw;
      hbf[erow * HH + col] = (__bf16)hw;
    }
    __syncthreads();  // h updated + gib[cur] free for re-targeting at t+1
  }

  const float inv = 1.f / (float)TT;
  for (int j = 0; j < 8; ++j)
    pooled[(size_t)(b0 + erow) * HH + ej0 + j] = acc[j] * inv;
}

// Phase 3: tiny projection, f32 scalar (1M FLOP, irrelevant to runtime)
__global__ __launch_bounds__(256) void proj_kernel(
    const float* __restrict__ pooled, const float* __restrict__ w_proj,
    const float* __restrict__ b_proj, float* __restrict__ out) {
  const int idx = blockIdx.x * 256 + threadIdx.x;
  if (idx >= BB * EE) return;
  const int b = idx / EE, e = idx % EE;
  const float* p = pooled + (size_t)b * HH;
  const float* w = w_proj + (size_t)e * HH;
  float s = b_proj[e];
  for (int k = 0; k < HH; ++k) s += p[k] * w[k];
  out[idx] = s;
}

extern "C" void kernel_launch(void* const* d_in, const int* in_sizes, int n_in,
                              void* d_out, int out_size, void* d_ws, size_t ws_size,
                              hipStream_t stream) {
  const float* x      = (const float*)d_in[0];
  const float* w_ih   = (const float*)d_in[1];
  const float* w_hh   = (const float*)d_in[2];
  const float* b_ih   = (const float*)d_in[3];
  const float* b_hh   = (const float*)d_in[4];
  const float* w_proj = (const float*)d_in[5];
  const float* b_proj = (const float*)d_in[6];

  // Workspace: gi [T,B,3H] f32 (192 MB) + pooled [B,H] f32
  float* gi     = (float*)d_ws;
  float* pooled = (float*)((char*)d_ws + (size_t)TT * BB * GG * sizeof(float));

  const int numWG = 512;
  gi_gemm_kernel<<<numWG, 256, 0, stream>>>(x, w_ih, b_ih, gi, numWG);
  gru_scan_kernel<<<4, 256, 0, stream>>>(gi, w_hh, b_hh, pooled);
  proj_kernel<<<(BB * EE + 255) / 256, 256, 0, stream>>>(pooled, w_proj, b_proj,
                                                         (float*)d_out);
}
